// HungarianMatcher_83648783057218
// MI455X (gfx1250) — matrix operand order repaired and checked by reading, not compiled, not run on hardware
//
#include <hip/hip_runtime.h>
#include <hip/hip_bf16.h>
#include <math.h>

// ---------------- problem constants ----------------
#define BB 2
#define QQ 100
#define QP 112            // Q padded to 7 WMMA tiles
#define MM 64
#define NCLS 81
#define HHWW 147456       // 384*384
#define NP 1024
#define KCHUNK 1152       // per-workgroup K slice of HW (147456/1152 = 128 chunks)
#define KSTEP 32          // bf16 WMMA K
#define ASTRIDE 40        // LDS row stride (bf16 elems), padded vs 32
#define FSTRIDE 68        // LDS row stride (f32 elems) for BCE gemm, padded vs 64

typedef __attribute__((ext_vector_type(16))) __bf16 v16bf;
typedef __attribute__((ext_vector_type(8)))  float  v8f;
typedef __attribute__((ext_vector_type(2)))  float  v2f;

// ---------------------------------------------------------------------------
// Kernel 1: cost_class  = -softmax(pred_logits)[:, gt_labels]
// grid (Q, B) x 128
// ---------------------------------------------------------------------------
__global__ __launch_bounds__(128) void class_cost_kernel(
    const float* __restrict__ logits, const int* __restrict__ labels,
    float* __restrict__ wsClass)
{
    int b = blockIdx.y, q = blockIdx.x, tid = threadIdx.x;
    __shared__ float l[NCLS];
    __shared__ float red[128];
    const float* L = logits + (size_t)(b * QQ + q) * NCLS;
    if (tid < NCLS) l[tid] = L[tid];
    __syncthreads();

    float v = (tid < NCLS) ? l[tid] : -INFINITY;
    red[tid] = v;
    __syncthreads();
    for (int s = 64; s > 0; s >>= 1) {
        if (tid < s) red[tid] = fmaxf(red[tid], red[tid + s]);
        __syncthreads();
    }
    float mx = red[0];
    __syncthreads();
    float e = (tid < NCLS) ? __expf(l[tid] - mx) : 0.f;
    red[tid] = e;
    __syncthreads();
    for (int s = 64; s > 0; s >>= 1) {
        if (tid < s) red[tid] += red[tid + s];
        __syncthreads();
    }
    float denom = red[0];
    if (tid < MM) {
        int lab = labels[b * MM + tid];
        wsClass[(size_t)(b * QQ + q) * MM + tid] = -__expf(l[lab] - mx) / denom;
    }
}

// ---------------------------------------------------------------------------
// Kernel 2: gather sampled points.  grid (Q+M, B) x 256
//   r <  Q : xPts[b][q][p] = pred_masks[b][q][idx[p]]; pos[b][q] = mean softplus
//   r >= Q : yPts[b][m][p] = gt[b][m][idx[p]]
// ---------------------------------------------------------------------------
__global__ __launch_bounds__(256) void gather_kernel(
    const float* __restrict__ pred, const int* __restrict__ gt,
    const int* __restrict__ pidx,
    float* __restrict__ xPts, float* __restrict__ yPts,
    float* __restrict__ wsPos)
{
    int b = blockIdx.y, r = blockIdx.x, tid = threadIdx.x;
    __shared__ float red[256];
    if (r < QQ) {
        const float* src = pred + (size_t)(b * QQ + r) * HHWW;
        float local = 0.f;
        for (int p = tid; p < NP; p += 256) {
            float x = src[pidx[p]];
            xPts[(size_t)(b * QQ + r) * NP + p] = x;
            local += fmaxf(x, 0.f) + log1pf(__expf(-fabsf(x)));
        }
        red[tid] = local;
        __syncthreads();
        for (int s = 128; s > 0; s >>= 1) {
            if (tid < s) red[tid] += red[tid + s];
            __syncthreads();
        }
        if (tid == 0) wsPos[b * QQ + r] = red[0] * (1.f / NP);
    } else {
        int m = r - QQ;
        const int* src = gt + (size_t)(b * MM + m) * HHWW;
        for (int p = tid; p < NP; p += 256)
            yPts[(size_t)(b * MM + m) * NP + p] = (float)src[pidx[p]];
    }
}

// ---------------------------------------------------------------------------
// Kernel 3: XY[b][q][m] = (x . y)/NP  via V_WMMA_F32_16X16X4_F32 (exact f32)
// grid B x 224 (7 waves; wave w owns m-tile row w, 4 n-tiles)
// ---------------------------------------------------------------------------
__global__ __launch_bounds__(224) void bce_gemm_kernel(
    const float* __restrict__ xPts, const float* __restrict__ yPts,
    float* __restrict__ wsXY)
{
    __shared__ float ldsX[QP * FSTRIDE];
    __shared__ float ldsY[MM * FSTRIDE];
    int b = blockIdx.x, tid = threadIdx.x;
    int wave = tid >> 5, lane = tid & 31, hf = lane >> 4, lr = lane & 15;

    int xRow = tid >> 1;                  // 0..111
    int colBase = (tid & 1) * 32;         // 0 or 32
    v8f acc[4] = {};

    // rows >= Q never written again: zero once
    if (xRow >= QQ) {
        #pragma unroll
        for (int i = 0; i < 32; ++i) ldsX[xRow * FSTRIDE + colBase + i] = 0.f;
    }

    for (int kb = 0; kb < NP; kb += 64) {
        if (xRow < QQ) {
            const float* src = xPts + (size_t)(b * QQ + xRow) * NP + kb + colBase;
            #pragma unroll
            for (int i = 0; i < 32; ++i) ldsX[xRow * FSTRIDE + colBase + i] = src[i];
        }
        if (tid < 128) {
            int yRow = tid >> 1;
            const float* src = yPts + (size_t)(b * MM + yRow) * NP + kb + colBase;
            #pragma unroll
            for (int i = 0; i < 32; ++i) ldsY[yRow * FSTRIDE + colBase + i] = src[i];
        }
        __syncthreads();
        if (wave < 7) {
            #pragma unroll
            for (int ks = 0; ks < 16; ++ks) {          // 16 steps of K=4
                v2f a;
                a[0] = ldsX[(wave * 16 + lr) * FSTRIDE + ks * 4 + hf * 2 + 0];
                a[1] = ldsX[(wave * 16 + lr) * FSTRIDE + ks * 4 + hf * 2 + 1];
                #pragma unroll
                for (int c = 0; c < 4; ++c) {
                    v2f bf;
                    bf[0] = ldsY[(c * 16 + lr) * FSTRIDE + ks * 4 + hf * 2 + 0];
                    bf[1] = ldsY[(c * 16 + lr) * FSTRIDE + ks * 4 + hf * 2 + 1];
                    acc[c] = __builtin_amdgcn_wmma_f32_16x16x4_f32(
                        false, a, false, bf, (short)0, acc[c], false, false);
                }
            }
        }
        __syncthreads();
    }
    if (wave < 7) {
        #pragma unroll
        for (int c = 0; c < 4; ++c)
            #pragma unroll
            for (int v = 0; v < 8; ++v) {
                int q = wave * 16 + hf * 8 + v;
                if (q < QQ)
                    wsXY[(size_t)(b * QQ + q) * MM + c * 16 + lr] =
                        acc[c][v] * (1.f / NP);
            }
    }
}

// ---------------------------------------------------------------------------
// Kernel 4: main einsum  I[b][q][m] = sum_h sigmoid(pred) * gt   (bf16 WMMA)
// Double-buffered LDS, one barrier per 32-wide K step; global prefetch of the
// stage-after-next.  Fused pred_sum / gt_sum.  grid (HW/KCHUNK, B) x 224.
// ---------------------------------------------------------------------------
__global__ __launch_bounds__(224) void intersect_kernel(
    const float* __restrict__ pred, const int* __restrict__ gt,
    float* __restrict__ wsI, float* __restrict__ wsPS, float* __restrict__ wsGS)
{
    __shared__ __bf16 ldsA[2][QP * ASTRIDE];
    __shared__ __bf16 ldsB[2][MM * ASTRIDE];
    int b = blockIdx.y;
    int k0 = blockIdx.x * KCHUNK;
    int tid = threadIdx.x;
    int wave = tid >> 5, lane = tid & 31, hf = lane >> 4, lr = lane & 15;

    // A loader: thread owns row tid/2, cols (tid&1)*16 .. +16 of each 32-step
    int aRow = tid >> 1;                      // 0..111
    int cb = (tid & 1) * 16;
    const float* aPtr = (aRow < QQ) ? pred + (size_t)(b * QQ + aRow) * HHWW : nullptr;
    // B loader: threads 0..127, row tid/2
    const int* gPtr = (tid < 128) ? gt + (size_t)(b * MM + (tid >> 1)) * HHWW : nullptr;

    if (!aPtr) {   // pad rows: written once per buffer, never touched again
        #pragma unroll
        for (int i = 0; i < 16; ++i) {
            ldsA[0][aRow * ASTRIDE + cb + i] = (__bf16)0.f;
            ldsA[1][aRow * ASTRIDE + cb + i] = (__bf16)0.f;
        }
    }

    float psAcc = 0.f, gsAcc = 0.f;
    v8f acc[4] = {};

    // ---- stage loader (global -> sigmoid/convert -> LDS[buf]) ----
    auto load_stage = [&](int bufSel, int kk) {
        if (aPtr) {
            if (kk + KSTEP + cb + 16 <= HHWW)
                __builtin_prefetch(aPtr + kk + KSTEP + cb, 0, 1);
            #pragma unroll
            for (int i = 0; i < 16; ++i) {
                float v = aPtr[kk + cb + i];
                float s = 1.f / (1.f + __expf(-v));   // sigmoid
                psAcc += s;
                ldsA[bufSel][aRow * ASTRIDE + cb + i] = (__bf16)s;
            }
        }
        if (gPtr) {
            if (kk + KSTEP + cb + 16 <= HHWW)
                __builtin_prefetch(gPtr + kk + KSTEP + cb, 0, 1);
            #pragma unroll
            for (int i = 0; i < 16; ++i) {
                float g = (float)gPtr[kk + cb + i];
                gsAcc += g;
                ldsB[bufSel][(tid >> 1) * ASTRIDE + cb + i] = (__bf16)g;
            }
        }
    };

    load_stage(0, k0);            // prologue
    int buf = 0;

    for (int ks = 0; ks < KCHUNK; ks += KSTEP) {
        __syncthreads();          // stores to ldsX[buf] visible; prior reads done
        if (ks + KSTEP < KCHUNK)
            load_stage(buf ^ 1, k0 + ks + KSTEP);   // overlap with WMMA below
        if (wave < 7) {
            // A fragment: 16-bit A 16x32 layout, k = (j/8)*16 + half*8 + j%8
            v16bf afrag;
            #pragma unroll
            for (int j = 0; j < 16; ++j) {
                int k = ((j >> 3) << 4) + (hf << 3) + (j & 7);
                afrag[j] = ldsA[buf][(wave * 16 + lr) * ASTRIDE + k];
            }
            #pragma unroll
            for (int c = 0; c < 4; ++c) {
                // B fragment: 16-bit B 32x16 layout, k = (j/8)*16 + half*8 + j%8
                v16bf bfrag;
                #pragma unroll
                for (int j = 0; j < 16; ++j)
                    bfrag[j] = ldsB[buf][(c * 16 + lr) * ASTRIDE + ((j >> 3) << 4) + (hf << 3) + (j & 7)];
                acc[c] = __builtin_amdgcn_wmma_f32_16x16x32_bf16(
                    false, afrag, false, bfrag, (short)0, acc[c], false, false);
            }
        }
        buf ^= 1;
    }

    if (aPtr) atomicAdd(&wsPS[b * QQ + aRow], psAcc);
    if (gPtr) atomicAdd(&wsGS[b * MM + (tid >> 1)], gsAcc);

    if (wave < 7) {
        #pragma unroll
        for (int c = 0; c < 4; ++c)
            #pragma unroll
            for (int v = 0; v < 8; ++v) {
                int q = wave * 16 + hf * 8 + v;
                if (q < QQ)
                    atomicAdd(&wsI[(size_t)(b * QQ + q) * MM + c * 16 + lr],
                              acc[c][v]);
            }
    }
}

// ---------------------------------------------------------------------------
// Kernel 5: combine.  grid B*Q x 64
// ---------------------------------------------------------------------------
__global__ __launch_bounds__(64) void combine_kernel(
    const float* __restrict__ wsI, const float* __restrict__ wsPS,
    const float* __restrict__ wsGS, const float* __restrict__ wsXY,
    const float* __restrict__ wsClass, const float* __restrict__ wsPos,
    float* __restrict__ out)
{
    int bq = blockIdx.x, m = threadIdx.x;
    int b = bq / QQ;
    float I    = wsI[(size_t)bq * MM + m];
    float dice = 1.f - 2.f * I / (wsPS[bq] + wsGS[b * MM + m] + 1e-6f);
    float bce  = wsPos[bq] - wsXY[(size_t)bq * MM + m];
    float cls  = wsClass[(size_t)bq * MM + m];
    float c = 2.f * cls + 5.f * bce + 5.f * dice;
    if (isnan(c))      c = 1.0e6f;
    else if (isinf(c)) c = (c > 0.f) ? 1.0e6f : -1.0e6f;
    out[(size_t)bq * MM + m] = c;
}

// ---------------------------------------------------------------------------
// launch
// ---------------------------------------------------------------------------
extern "C" void kernel_launch(void* const* d_in, const int* in_sizes, int n_in,
                              void* d_out, int out_size, void* d_ws, size_t ws_size,
                              hipStream_t stream) {
    const float* pred_logits = (const float*)d_in[0];
    const float* pred_masks  = (const float*)d_in[1];
    const int*   gt_labels   = (const int*)d_in[2];
    const int*   gt_masks    = (const int*)d_in[3];
    const int*   point_idx   = (const int*)d_in[4];
    (void)in_sizes; (void)n_in; (void)out_size; (void)ws_size;

    float* ws      = (float*)d_ws;
    float* wsI     = ws;                       // B*Q*M = 12800
    float* wsPS    = ws + 12800;               // B*Q   = 200
    float* wsGS    = ws + 13000;               // B*M   = 128
    float* wsXY    = ws + 13128;               // B*Q*M = 12800
    float* wsClass = ws + 25928;               // B*Q*M = 12800
    float* wsPos   = ws + 38728;               // B*Q   = 200
    float* xPts    = ws + 38928;               // B*Q*NP = 204800
    float* yPts    = ws + 243728;              // B*M*NP = 131072

    // zero the atomic accumulators (I, predSum, gtSum are contiguous)
    hipMemsetAsync(wsI, 0, (size_t)(12800 + 200 + 128) * sizeof(float), stream);

    class_cost_kernel<<<dim3(QQ, BB), 128, 0, stream>>>(pred_logits, gt_labels, wsClass);
    gather_kernel<<<dim3(QQ + MM, BB), 256, 0, stream>>>(pred_masks, gt_masks,
                                                         point_idx, xPts, yPts, wsPos);
    bce_gemm_kernel<<<BB, 224, 0, stream>>>(xPts, yPts, wsXY);
    intersect_kernel<<<dim3(HHWW / KCHUNK, BB), 224, 0, stream>>>(
        pred_masks, gt_masks, wsI, wsPS, wsGS);
    combine_kernel<<<BB * QQ, 64, 0, stream>>>(wsI, wsPS, wsGS, wsXY, wsClass,
                                               wsPos, (float*)d_out);
}
